// GIN_37658273251987
// MI455X (gfx1250) — matrix operand order repaired and checked by reading, not compiled, not run on hardware
//
#include <hip/hip_runtime.h>
#include <hip/hip_bf16.h>

// ---------------- problem constants ----------------
#define Nn   25000
#define Ee   500000
#define Cc   128
#define DIMd 512
#define OUTo 256
#define CAT  (2 * DIMd + OUTo)   // 1280
#define BN_EPS 1e-5f

typedef __attribute__((ext_vector_type(16))) __bf16 v16bf;
typedef __attribute__((ext_vector_type(8)))  __bf16 v8bf;
typedef __attribute__((ext_vector_type(4)))  __bf16 v4bf;
typedef __attribute__((ext_vector_type(8)))  float  v8f;
typedef int v4i_ __attribute__((vector_size(4 * sizeof(int))));

// Async global->LDS path (gfx1250): guarded so either toolchain compiles.
#if defined(__gfx1250__) && \
    __has_builtin(__builtin_amdgcn_global_load_async_to_lds_b128) && \
    __has_builtin(__builtin_amdgcn_s_wait_asynccnt)
#define USE_ASYNC_LDS 1
#else
#define USE_ASYNC_LDS 0
#endif

#define SHUF16(lo, hi) __builtin_shufflevector(lo, hi, 0,1,2,3,4,5,6,7,8,9,10,11,12,13,14,15)

// =====================================================================
// fp32 -> bf16 bulk conversion (weights once, edge_attr once per launch)
// =====================================================================
__global__ void cvt_bf16_kernel(const float* __restrict__ in, __bf16* __restrict__ out, size_t n) {
    size_t i = ((size_t)blockIdx.x * blockDim.x + threadIdx.x) * 4;
    size_t stride = (size_t)gridDim.x * blockDim.x * 4;
    for (; i < n; i += stride) {
        float4 f = *(const float4*)(in + i);
        v4bf b = { (__bf16)f.x, (__bf16)f.y, (__bf16)f.z, (__bf16)f.w };
        *(v4bf*)(out + i) = b;
    }
}

// =====================================================================
// BatchNorm statistics: one block per channel, 256-thread reduction.
// stats[c] = mean, stats[C+c] = rsqrt(var + eps)
// =====================================================================
__global__ void bn_stats_kernel(const float* __restrict__ X, float* __restrict__ stats) {
    const int c = blockIdx.x;
    const int t = threadIdx.x;
    float s = 0.f, s2 = 0.f;
    for (int n = t; n < Nn; n += 256) {
        float v = X[(size_t)n * Cc + c];
        s += v; s2 += v * v;
    }
    __shared__ float sh[256], sh2[256];
    sh[t] = s; sh2[t] = s2;
    __syncthreads();
    for (int off = 128; off > 0; off >>= 1) {
        if (t < off) { sh[t] += sh[t + off]; sh2[t] += sh2[t + off]; }
        __syncthreads();
    }
    if (t == 0) {
        float mean = sh[0] / (float)Nn;
        float var  = sh2[0] / (float)Nn - mean * mean;
        stats[c]      = mean;
        stats[Cc + c] = rsqrtf(var + BN_EPS);
    }
}

__global__ void bn_apply_kernel(const float* __restrict__ X, const float* __restrict__ stats,
                                const float* __restrict__ gamma, const float* __restrict__ beta,
                                float* __restrict__ x0) {
    int i = blockIdx.x * 256 + threadIdx.x;
    if (i < Nn * Cc) {
        int c = i & (Cc - 1);
        x0[i] = (X[i] - stats[c]) * stats[Cc + c] * gamma[c] + beta[c];
    }
}

__global__ void zero_kernel(float* __restrict__ p, size_t n) {
    size_t i = (size_t)blockIdx.x * blockDim.x + threadIdx.x;
    size_t stride = (size_t)gridDim.x * blockDim.x;
    for (; i < n; i += stride) p[i] = 0.f;
}

// =====================================================================
// Fused GINE edge kernel (bf16 inputs, f32 accumulate):
//   msg = ReLU(xin[src] + edge_attr @ W^T + b)   -> atomic scatter to aggr[dst]
// One block = 16 edges; A tile async-staged to LDS; A fragments hoisted
// into registers; all 4 B fragments preloaded so the 4 WMMAs issue
// back-to-back; xin gather issued before the WMMA chain to hide latency.
// =====================================================================
__global__ void edge_conv_kernel(const __bf16* __restrict__ edge_bf,  // [E, 128]
                                 const __bf16* __restrict__ Wbf,      // [Fout, 128]
                                 const float*  __restrict__ bias,
                                 const float*  __restrict__ xin,      // [N, Fout]
                                 const int*    __restrict__ src,
                                 const int*    __restrict__ dst,
                                 float*        __restrict__ aggr,     // [N, Fout]
                                 int Fout) {
    __shared__ __align__(16) __bf16 As[16 * Cc];   // 4 KB bf16 tile
    __shared__ int s_src[16], s_dst[16];

    const int tid = threadIdx.x;
    const size_t e0 = (size_t)blockIdx.x * 16;

    // ---- stage 16x128 bf16 edge tile: 256 threads x 16B ----
    const __bf16* gsrc = edge_bf + e0 * Cc + tid * 8;
#if USE_ASYNC_LDS
    __builtin_amdgcn_global_load_async_to_lds_b128(
        (__attribute__((address_space(1))) v4i_*)gsrc,
        (__attribute__((address_space(3))) v4i_*)(As + tid * 8),
        0, 0);
#else
    *(v8bf*)(As + tid * 8) = *(const v8bf*)gsrc;
#endif
    __builtin_prefetch(edge_bf + (e0 + 16) * Cc + tid * 8, 0, 1);  // warm next tile
    if (tid < 16)      s_src[tid]      = src[e0 + tid];
    else if (tid < 32) s_dst[tid - 16] = dst[e0 + tid - 16];
#if USE_ASYNC_LDS
    __builtin_amdgcn_s_wait_asynccnt(0);
#endif
    __syncthreads();

    const int wave = tid >> 5, lane = tid & 31;
    const int h = lane >> 4, ln = lane & 15;

    // ---- A fragments (16x32 bf16 layout), built once, reused for all ct ----
    v16bf afrag[4];
    #pragma unroll
    for (int j = 0; j < 4; ++j) {
        const __bf16* ap = As + ln * Cc + j * 32 + h * 8;
        v8bf lo = *(const v8bf*)ap;
        v8bf hi = *(const v8bf*)(ap + 16);
        afrag[j] = SHUF16(lo, hi);
    }

    const int nColTiles = Fout >> 4;
    for (int ct = wave; ct < nColTiles; ct += 8) {
        const int ob  = ct << 4;
        const int col = ob + ln;

        // preload all 4 B fragments (one clause of 8 b128 loads)
        v16bf bfrag[4];
        const __bf16* wbase = Wbf + (size_t)col * Cc;
        #pragma unroll
        for (int j = 0; j < 4; ++j) {
            const __bf16* wp = wbase + j * 32 + h * 8;
            bfrag[j] = SHUF16(*(const v8bf*)wp, *(const v8bf*)(wp + 16));
        }

        // issue the xin gather early so it overlaps the WMMA chain
        float xv[8];
        #pragma unroll
        for (int v = 0; v < 8; ++v)
            xv[v] = xin[(size_t)s_src[v + h * 8] * Fout + col];

        v8f acc = {};
        #pragma unroll
        for (int j = 0; j < 4; ++j)
            acc = __builtin_amdgcn_wmma_f32_16x16x32_bf16(false, afrag[j], false, bfrag[j],
                                                          (short)0, acc, false, false);

        // ---- epilogue: + bias + xin[src], ReLU, scatter-add to aggr[dst] ----
        const float bv = bias[col];
        #pragma unroll
        for (int v = 0; v < 8; ++v) {
            float val = fmaxf(acc[v] + bv + xv[v], 0.f);
            atomicAdd(aggr + (size_t)s_dst[v + h * 8] * Fout + col, val);
        }
    }
}

// =====================================================================
// Node GEMM + tanh:  out = tanh((A1 [+ A2]) @ W^T + b)
// Single 16-row staging per block; 8 waves loop over all column tiles.
// K unrolled by 128: 4 A-frag + 4 B-frag loads grouped, then 4 WMMAs.
// =====================================================================
__global__ void node_gemm_kernel(const float* __restrict__ A1,
                                 const float* __restrict__ A2,     // may be null
                                 const __bf16* __restrict__ Wbf,   // [Fout, K] bf16
                                 const float* __restrict__ bias,   // may be null
                                 float* __restrict__ out_ws,       // may be null
                                 float* __restrict__ out_cat,
                                 int K, int Fout, int catOff) {
    __shared__ __align__(16) __bf16 As[16 * DIMd];   // up to 16 KB
    const int tid  = threadIdx.x;
    const int row0 = blockIdx.x * 16;

    // ---- stage (A1 [+A2]) 16xK fp32 -> bf16 LDS, zero-fill past N ----
    for (int base = tid * 4; base < 16 * K; base += 256 * 4) {
        int r = base / K, c = base % K;
        int row = row0 + r;
        float4 f = make_float4(0.f, 0.f, 0.f, 0.f);
        if (row < Nn) {
            f = *(const float4*)(A1 + (size_t)row * K + c);
            if (A2) {
                float4 g = *(const float4*)(A2 + (size_t)row * K + c);
                f.x += g.x; f.y += g.y; f.z += g.z; f.w += g.w;
            }
        }
        v4bf b = { (__bf16)f.x, (__bf16)f.y, (__bf16)f.z, (__bf16)f.w };
        *(v4bf*)(As + r * K + c) = b;
    }
    __syncthreads();

    const int wave = tid >> 5, lane = tid & 31;
    const int h = lane >> 4, ln = lane & 15;
    const int nColTiles = Fout >> 4;

    for (int ct = wave; ct < nColTiles; ct += 8) {
        const int ob  = ct << 4;
        const int col = ob + ln;
        v8f acc = {};
        for (int kk = 0; kk < K; kk += 128) {
            v16bf a4[4], b4[4];
            #pragma unroll
            for (int j = 0; j < 4; ++j) {
                const __bf16* ap = As + ln * K + kk + j * 32 + h * 8;
                a4[j] = SHUF16(*(const v8bf*)ap, *(const v8bf*)(ap + 16));
                const __bf16* wp = Wbf + (size_t)col * K + kk + j * 32 + h * 8;
                b4[j] = SHUF16(*(const v8bf*)wp, *(const v8bf*)(wp + 16));
            }
            #pragma unroll
            for (int j = 0; j < 4; ++j)
                acc = __builtin_amdgcn_wmma_f32_16x16x32_bf16(false, a4[j], false, b4[j],
                                                              (short)0, acc, false, false);
        }
        const float bv = bias ? bias[col] : 0.f;
        #pragma unroll
        for (int v = 0; v < 8; ++v) {
            const int row = row0 + v + h * 8;
            if (row < Nn) {
                float val = tanhf(acc[v] + bv);
                if (out_ws) out_ws[(size_t)row * Fout + col] = val;
                out_cat[(size_t)row * CAT + catOff + col] = val;
            }
        }
    }
}

// =====================================================================
// Host-side launcher
// =====================================================================
extern "C" void kernel_launch(void* const* d_in, const int* in_sizes, int n_in,
                              void* d_out, int out_size, void* d_ws, size_t ws_size,
                              hipStream_t stream) {
    (void)in_sizes; (void)n_in; (void)out_size; (void)ws_size;

    const float* X         = (const float*)d_in[0];
    const float* edge_attr = (const float*)d_in[1];
    const float* bn_gamma  = (const float*)d_in[2];
    const float* bn_beta   = (const float*)d_in[3];
    const float* lin1_W    = (const float*)d_in[4];
    const float* lin1_b    = (const float*)d_in[5];
    const float* nn1_W     = (const float*)d_in[6];
    const float* nn1_b     = (const float*)d_in[7];
    const float* lin2_W    = (const float*)d_in[8];
    const float* lin2_b    = (const float*)d_in[9];
    const float* nn2_W     = (const float*)d_in[10];
    const float* nn2_b     = (const float*)d_in[11];
    const float* fc1_W     = (const float*)d_in[12];
    const int*   edge_idx  = (const int*)d_in[13];
    const int*   src = edge_idx;
    const int*   dst = edge_idx + Ee;

    float* out = (float*)d_out;

    // ---- workspace layout ----
    float* ws    = (float*)d_ws;
    float* x0    = ws;                                  // [N, C]
    float* aggr1 = x0    + (size_t)Nn * Cc;             // [N, C]
    float* x1    = aggr1 + (size_t)Nn * Cc;             // [N, DIM]
    float* aggr2 = x1    + (size_t)Nn * DIMd;           // [N, DIM]
    float* x2    = aggr2 + (size_t)Nn * DIMd;           // [N, DIM]
    float* stats = x2    + (size_t)Nn * DIMd;           // [2*C]

    __bf16* lin1bf = (__bf16*)(stats + 2 * Cc);         // [C,   C]
    __bf16* nn1bf  = lin1bf + (size_t)Cc * Cc;          // [DIM, C]
    __bf16* lin2bf = nn1bf  + (size_t)DIMd * Cc;        // [DIM, C]
    __bf16* nn2bf  = lin2bf + (size_t)DIMd * Cc;        // [DIM, DIM]
    __bf16* fc1bf  = nn2bf  + (size_t)DIMd * DIMd;      // [OUT, DIM]
    __bf16* edgebf = fc1bf  + (size_t)OUTo * DIMd;      // [E, C]

    const int rowTiles  = (Nn + 15) / 16;    // 1563
    const int edgeTiles = Ee / 16;           // 31250

    // 0) bulk fp32 -> bf16 conversions (weights + edge features)
    cvt_bf16_kernel<<<64,   256, 0, stream>>>(lin1_W, lin1bf, (size_t)Cc * Cc);
    cvt_bf16_kernel<<<64,   256, 0, stream>>>(nn1_W,  nn1bf,  (size_t)DIMd * Cc);
    cvt_bf16_kernel<<<64,   256, 0, stream>>>(lin2_W, lin2bf, (size_t)DIMd * Cc);
    cvt_bf16_kernel<<<256,  256, 0, stream>>>(nn2_W,  nn2bf,  (size_t)DIMd * DIMd);
    cvt_bf16_kernel<<<128,  256, 0, stream>>>(fc1_W,  fc1bf,  (size_t)OUTo * DIMd);
    cvt_bf16_kernel<<<4096, 256, 0, stream>>>(edge_attr, edgebf, (size_t)Ee * Cc);

    // 1) BatchNorm
    bn_stats_kernel<<<Cc, 256, 0, stream>>>(X, stats);
    bn_apply_kernel<<<(Nn * Cc + 255) / 256, 256, 0, stream>>>(X, stats, bn_gamma, bn_beta, x0);

    // 2) zero aggregation buffers
    zero_kernel<<<2048, 256, 0, stream>>>(aggr1, (size_t)Nn * Cc);
    zero_kernel<<<2048, 256, 0, stream>>>(aggr2, (size_t)Nn * DIMd);

    // 3) GINEConv1 -> x1 (d_out cols [0,512))
    edge_conv_kernel<<<edgeTiles, 256, 0, stream>>>(edgebf, lin1bf, lin1_b,
                                                    x0, src, dst, aggr1, Cc);
    node_gemm_kernel<<<rowTiles, 256, 0, stream>>>(x0, aggr1, nn1bf, nn1_b,
                                                   x1, out, Cc, DIMd, 0);

    // 4) GINEConv2 -> x2 (d_out cols [512,1024))
    edge_conv_kernel<<<edgeTiles, 256, 0, stream>>>(edgebf, lin2bf, lin2_b,
                                                    x1, src, dst, aggr2, DIMd);
    node_gemm_kernel<<<rowTiles, 256, 0, stream>>>(x1, aggr2, nn2bf, nn2_b,
                                                   x2, out, DIMd, DIMd, 512);

    // 5) fc1 + tanh -> x3 (d_out cols [1024,1280))
    node_gemm_kernel<<<rowTiles, 256, 0, stream>>>(x2, (const float*)nullptr, fc1bf,
                                                   (const float*)nullptr,
                                                   (float*)nullptr, out, DIMd, OUTo, 1024);
}
